// Se3NN_78271484002961
// MI455X (gfx1250) — matrix operand order repaired and checked by reading, not compiled, not run on hardware
//
#include <hip/hip_runtime.h>
#include <cmath>
#include <cstdint>
#include <cstddef>

#define DMAX   112
#define NSC    32
#define NPROT  20000
#define NPHARM 5000
#define EPP    30000
#define EHH    15000
#define NGRAPH 64
#define RDIMC  50
#define MAXRC  6.0f

typedef __attribute__((ext_vector_type(16))) _Float16 v16h;
typedef __attribute__((ext_vector_type(8)))  _Float16 v8h;
typedef __attribute__((ext_vector_type(8)))  float    v8f;

struct TPPath { int ioff, m1, l1, l2, ooff, mo, lo, woff; float norm; };
struct TPParams { int np; TPPath p[8]; };
struct BNIr { int off, m, l; };
struct BNParams { int nir; BNIr ir[4]; };

// ---------------------------------------------------------------------------
// f16-WMMA GEMM with LDS tile staging: C[M,N] = act(A[M,K] @ B[K,N] + bias).
// Block = 128 threads (4 waves); K-step 32.
//   nsub=1: 16(M) x 64(N) tile, one 16x16 subtile per wave.
//   nsub=2: 16(M) x 128(N) tile, two subtiles per wave (one A-frag feeds two
//           WMMAs -> half the barriers / A re-staging for the wide W2 GEMMs).
// A (16x32) and B (32 x colspan, transposed to n-major) staged in LDS as f16
// so WMMA fragment reads are unconditional 16B LDS vector loads. Stride 40
// halfwords = 80B keeps 16B alignment and distinct banks across 16 lanes.
// Optional row-gather on A (gatherA) and pre-activation ReLU on A.
// ---------------------------------------------------------------------------
#define AST 40
#define BST 40

__global__ __launch_bounds__(128) void wmma_gemm_kernel(
    const float* __restrict__ A, int lda,
    const float* __restrict__ B,   // [K,N] row-major
    const float* __restrict__ bias,
    float* __restrict__ C, int ldc,
    int M, int N, int K,
    const int* __restrict__ gatherA,
    int preReluA, int reluOut, int nsub)
{
    __shared__ __align__(16) _Float16 ldsA[16 * AST];
    __shared__ __align__(16) _Float16 ldsBt[128 * BST];

    int t = threadIdx.x;
    int wave = t >> 5, lane = t & 31;
    int half = lane >> 4, l16 = lane & 15;
    int colspan = nsub << 6;                 // 64 or 128
    int csh = (nsub == 2) ? 7 : 6;
    int rowBase  = blockIdx.x * 16;          // always < M by grid construction
    int colBase0 = blockIdx.y * colspan;
    int nloc0 = wave * (nsub << 4);          // wave's first local column
    bool waveActive = (colBase0 + nloc0 < N);  // wave-uniform

    // each thread stages 4 consecutive A elements of one row per K-step
    int arow = t >> 3;                       // 0..15
    int akb  = (t & 7) << 2;                 // 0,4,...,28
    int r = rowBase + arow;
    bool rowOK = (r < M);
    int ra = rowOK ? (gatherA ? gatherA[r] : r) : 0;
    const float* Arow = A + (size_t)ra * lda;

    v8f acc0 = {0.f,0.f,0.f,0.f,0.f,0.f,0.f,0.f};
    v8f acc1 = {0.f,0.f,0.f,0.f,0.f,0.f,0.f,0.f};

    for (int k0 = 0; k0 < K; k0 += 32) {
        // ---- stage A tile (16 rows x 32 k) as f16, pre-ReLU folded in ----
        {
            float v0 = 0.f, v1 = 0.f, v2 = 0.f, v3 = 0.f;
            if (rowOK) {
                if (k0 + 32 <= K) {
                    const float* p = Arow + k0 + akb;
                    v0 = p[0]; v1 = p[1]; v2 = p[2]; v3 = p[3];
                } else {
                    int k = k0 + akb;
                    if (k + 0 < K) v0 = Arow[k + 0];
                    if (k + 1 < K) v1 = Arow[k + 1];
                    if (k + 2 < K) v2 = Arow[k + 2];
                    if (k + 3 < K) v3 = Arow[k + 3];
                }
            }
            if (preReluA) {
                v0 = fmaxf(v0, 0.f); v1 = fmaxf(v1, 0.f);
                v2 = fmaxf(v2, 0.f); v3 = fmaxf(v3, 0.f);
            }
            _Float16* d = ldsA + arow * AST + akb;
            d[0] = (_Float16)v0; d[1] = (_Float16)v1;
            d[2] = (_Float16)v2; d[3] = (_Float16)v3;
        }
        // ---- stage B tile (32 k x colspan n), transposed to [n][k] in LDS ----
        {
            bool fastB = (k0 + 32 <= K) && (colBase0 + colspan <= N);
            int iters = colspan >> 2;        // 16 or 32
            for (int it = 0; it < iters; ++it) {
                int idx = t + (it << 7);     // coalesced across 128 threads
                int kk = idx >> csh;         // 0..31
                int nn = idx & (colspan - 1);
                float v;
                if (fastB) {
                    v = B[(size_t)(k0 + kk) * N + colBase0 + nn];
                } else {
                    int k = k0 + kk, n = colBase0 + nn;
                    v = (k < K && n < N) ? B[(size_t)k * N + n] : 0.f;
                }
                ldsBt[nn * BST + kk] = (_Float16)v;
            }
        }
        // ---- prefetch next K-step tiles (speculative, gfx1250 global_prefetch) ----
        if (k0 + 32 < K) {
            if (rowOK) __builtin_prefetch(Arow + k0 + 32 + akb, 0, 1);
            int pk = k0 + 32 + (t >> 2);
            if (pk < K) __builtin_prefetch(B + (size_t)pk * N + colBase0 + ((t & 3) << 4), 0, 1);
        }
        __syncthreads();
        if (waveActive) {   // wave-uniform: EXEC stays all-ones around WMMA
            // A frag: K(e) = half*16 + e -> one contiguous 16-halfword run
            const _Float16* pa = ldsA + l16 * AST + (half << 4);
            v8h a0 = *(const v8h*)pa;
            v8h a1 = *(const v8h*)(pa + 8);
            v16h af;
            #pragma unroll
            for (int e = 0; e < 8; ++e) { af[e] = a0[e]; af[8 + e] = a1[e]; }
            // B frag: K(e) = half*16 + e -> one contiguous 16-halfword run
            {
                const _Float16* pb = ldsBt + (nloc0 + l16) * BST + (half << 4);
                v8h b0 = *(const v8h*)pb;
                v8h b1 = *(const v8h*)(pb + 8);
                v16h bf;
                #pragma unroll
                for (int e = 0; e < 8; ++e) { bf[e] = b0[e]; bf[8 + e] = b1[e]; }
                acc0 = __builtin_amdgcn_wmma_f32_16x16x32_f16(false, af, false, bf,
                                                              (short)0, acc0, false, false);
            }
            if (nsub == 2) {
                const _Float16* pb = ldsBt + (nloc0 + 16 + l16) * BST + (half << 4);
                v8h b0 = *(const v8h*)pb;
                v8h b1 = *(const v8h*)(pb + 8);
                v16h bf;
                #pragma unroll
                for (int e = 0; e < 8; ++e) { bf[e] = b0[e]; bf[8 + e] = b1[e]; }
                acc1 = __builtin_amdgcn_wmma_f32_16x16x32_f16(false, af, false, bf,
                                                              (short)0, acc1, false, false);
            }
        }
        __syncthreads();
    }

    if (waveActive) {
        bool fullM = (rowBase + 16 <= M);    // block-uniform fast path
        {
            int col = colBase0 + nloc0 + l16;
            if (col < N) {
                float bv = bias ? bias[col] : 0.f;
                float* cp = C + (size_t)rowBase * ldc + col;
                if (fullM) {
                    #pragma unroll
                    for (int rr = 0; rr < 8; ++rr) {
                        float v = acc0[rr] + bv;
                        if (reluOut) v = v > 0.f ? v : 0.f;
                        cp[(size_t)(rr + (half << 3)) * ldc] = v;
                    }
                } else {
                    #pragma unroll
                    for (int rr = 0; rr < 8; ++rr) {
                        int row = rowBase + rr + (half << 3);
                        if (row < M) {
                            float v = acc0[rr] + bv;
                            if (reluOut) v = v > 0.f ? v : 0.f;
                            C[(size_t)row * ldc + col] = v;
                        }
                    }
                }
            }
        }
        if (nsub == 2) {
            int col = colBase0 + nloc0 + 16 + l16;
            if (col < N) {
                float bv = bias ? bias[col] : 0.f;
                float* cp = C + (size_t)rowBase * ldc + col;
                if (fullM) {
                    #pragma unroll
                    for (int rr = 0; rr < 8; ++rr) {
                        float v = acc1[rr] + bv;
                        if (reluOut) v = v > 0.f ? v : 0.f;
                        cp[(size_t)(rr + (half << 3)) * ldc] = v;
                    }
                } else {
                    #pragma unroll
                    for (int rr = 0; rr < 8; ++rr) {
                        int row = rowBase + rr + (half << 3);
                        if (row < M) {
                            float v = acc1[rr] + bv;
                            if (reluOut) v = v > 0.f ? v : 0.f;
                            C[(size_t)row * ldc + col] = v;
                        }
                    }
                }
            }
        }
    }
}

// ---------------------------------------------------------------------------
__global__ void fill_kernel(float* __restrict__ p, size_t n)
{
    size_t i = blockIdx.x * (size_t)blockDim.x + threadIdx.x;
    size_t st = (size_t)gridDim.x * blockDim.x;
    for (; i < n; i += st) p[i] = 0.f;
}

// sh (1, sqrt3*u) [+ reversed] and Gaussian radial smear per edge
__global__ void edge_geom_kernel(const float* __restrict__ attr, int E,
                                 float* __restrict__ sh, float* __restrict__ shrev,
                                 float* __restrict__ sm)
{
    int e = blockIdx.x * blockDim.x + threadIdx.x;
    if (e >= E) return;
    float x = attr[e*3+0], y = attr[e*3+1], z = attr[e*3+2];
    float n = sqrtf(x*x + y*y + z*z);
    float inv = 1.0f / fmaxf(n, 1e-9f);
    const float S3 = 1.7320508075688772f;
    float ux = S3*x*inv, uy = S3*y*inv, uz = S3*z*inv;
    sh[e*4+0] = 1.0f; sh[e*4+1] = ux; sh[e*4+2] = uy; sh[e*4+3] = uz;
    if (shrev) {
        shrev[e*4+0] = 1.0f; shrev[e*4+1] = -ux; shrev[e*4+2] = -uy; shrev[e*4+3] = -uz;
    }
    const float delta = MAXRC / (RDIMC - 1);
    const float coeff = -0.5f / (delta * delta);
    for (int r0 = 0; r0 < RDIMC; ++r0) {
        float t = n - delta * (float)r0;
        sm[(size_t)e * RDIMC + r0] = expf(coeff * t * t);
    }
}

// pf[ce,96] = concat(efeat[eg,0:32], fA[idxA[eg],0:32], fB[idxB[eg],0:32])
__global__ void build_feat_kernel(const float* __restrict__ efeat,
                                  const float* __restrict__ fA, const int* __restrict__ iA,
                                  const float* __restrict__ fB, const int* __restrict__ iB,
                                  float* __restrict__ pf, int e0, int ce)
{
    int t = blockIdx.x * blockDim.x + threadIdx.x;
    int e = t >> 5, c = t & 31;
    if (e >= ce) return;
    int eg = e0 + e;
    float* row = pf + (size_t)e * 96;
    row[c]      = efeat[(size_t)eg * NSC + c];
    row[32 + c] = fA[(size_t)iA[eg] * DMAX + c];
    row[64 + c] = fB[(size_t)iB[eg] * DMAX + c];
}

// fused tensor product + segment-sum scatter (one 32-lane group per edge)
__global__ void tp_scatter_kernel(const float* __restrict__ xsrc,
                                  const int* __restrict__ sidx,
                                  const int* __restrict__ didx,
                                  const float* __restrict__ sh,
                                  const float* __restrict__ wbuf,
                                  float* __restrict__ o,
                                  int wn, int e0, int ce, TPParams tp)
{
    int lane = threadIdx.x & 31;
    int el = (blockIdx.x * blockDim.x + threadIdx.x) >> 5;
    if (el >= ce) return;
    int eg = e0 + el;
    const float* xr  = xsrc + (size_t)sidx[eg] * DMAX;
    float*       orow = o   + (size_t)didx[eg] * DMAX;
    const float* shr = sh + (size_t)eg * 4;
    const float* wr  = wbuf + (size_t)el * wn;
    float y1 = shr[1], y2 = shr[2], y3 = shr[3];
    const float IS3 = 0.5773502691896258f;   // 1/sqrt(3)
    const float IS6 = 0.4082482904638631f;   // 1/sqrt(6)
    for (int pi = 0; pi < tp.np; ++pi) {
        TPPath P = tp.p[pi];
        int w = lane;
        if (w >= P.mo) continue;
        const float* wcol = wr + P.woff + w;
        if (P.l1 == 0) {
            float acc = 0.f;
            for (int u = 0; u < P.m1; ++u) acc += xr[P.ioff + u] * wcol[u * P.mo];
            if (P.l2 == 0) {
                atomicAdd(&orow[P.ooff + w], P.norm * acc);
            } else {
                float s = P.norm * IS3 * acc;
                float* op = &orow[P.ooff + w * 3];
                atomicAdd(op + 0, s * y1);
                atomicAdd(op + 1, s * y2);
                atomicAdd(op + 2, s * y3);
            }
        } else {
            float a0 = 0.f, a1 = 0.f, a2 = 0.f;
            for (int u = 0; u < P.m1; ++u) {
                float ww = wcol[u * P.mo];
                const float* xu = &xr[P.ioff + u * 3];
                float x0 = xu[0], x1 = xu[1], x2 = xu[2];
                if (P.l2 == 0) { a0 += x0 * ww; a1 += x1 * ww; a2 += x2 * ww; }
                else if (P.lo == 0) { a0 += (x0*y1 + x1*y2 + x2*y3) * ww; }
                else {  // l1=l2=lo=1: cross product via eps/sqrt6
                    a0 += (x1*y3 - x2*y2) * ww;
                    a1 += (x2*y1 - x0*y3) * ww;
                    a2 += (x0*y2 - x1*y1) * ww;
                }
            }
            if (P.l2 == 0) {
                float s = P.norm * IS3;
                float* op = &orow[P.ooff + w * 3];
                atomicAdd(op+0, s*a0); atomicAdd(op+1, s*a1); atomicAdd(op+2, s*a2);
            } else if (P.lo == 0) {
                atomicAdd(&orow[P.ooff + w], P.norm * IS3 * a0);
            } else {
                float s = P.norm * IS6;
                float* op = &orow[P.ooff + w * 3];
                atomicAdd(op+0, s*a0); atomicAdd(op+1, s*a1); atomicAdd(op+2, s*a2);
            }
        }
    }
}

__global__ void add_res_kernel(float* __restrict__ o, const float* __restrict__ res,
                               int N, int cmin)
{
    int t = blockIdx.x * blockDim.x + threadIdx.x;
    if (t >= N * cmin) return;
    int n = t / cmin, c = t % cmin;
    o[(size_t)n * DMAX + c] += res[(size_t)n * DMAX + c];
}

// one block per channel; deterministic LDS tree reduce (no atomics)
__global__ void bn_reduce_kernel(const float* __restrict__ o, int N, BNParams bn,
                                 float* __restrict__ stats)
{
    int ch = blockIdx.x;
    int off = 0, l = 0;
    {
        int base = 0;
        for (int i = 0; i < bn.nir; ++i) {
            if (ch < base + bn.ir[i].m) {
                int cin = ch - base; l = bn.ir[i].l;
                off = bn.ir[i].off + cin * (l ? 3 : 1);
                break;
            }
            base += bn.ir[i].m;
        }
    }
    __shared__ float s1[256], s2[256];
    float a = 0.f, b = 0.f;
    for (int n = threadIdx.x; n < N; n += blockDim.x) {
        const float* row = o + (size_t)n * DMAX + off;
        if (l == 0) { float x = row[0]; a += x; b += x * x; }
        else { float x0 = row[0], x1 = row[1], x2 = row[2]; b += x0*x0 + x1*x1 + x2*x2; }
    }
    s1[threadIdx.x] = a; s2[threadIdx.x] = b;
    __syncthreads();
    for (int st = 128; st > 0; st >>= 1) {
        if ((int)threadIdx.x < st) {
            s1[threadIdx.x] += s1[threadIdx.x + st];
            s2[threadIdx.x] += s2[threadIdx.x + st];
        }
        __syncthreads();
    }
    if (threadIdx.x == 0) { stats[ch] = s1[0]; stats[128 + ch] = s2[0]; }
}

__global__ void bn_apply_kernel(const float* __restrict__ o, int N, BNParams bn,
                                const float* __restrict__ stats,
                                float* __restrict__ dst, int dout, int accumulate)
{
    int n = blockIdx.x * blockDim.x + threadIdx.x;
    if (n >= N) return;
    const float* orow = o + (size_t)n * DMAX;
    float* drow = dst + (size_t)n * DMAX;
    if (!accumulate) for (int c = dout; c < DMAX; ++c) drow[c] = 0.f;
    float invN = 1.0f / (float)N;
    int ch = 0;
    for (int i = 0; i < bn.nir; ++i) {
        BNIr ir = bn.ir[i];
        if (ir.l == 0) {
            for (int c = 0; c < ir.m; ++c, ++ch) {
                float mean = stats[ch] * invN;
                float var = stats[128 + ch] * invN - mean * mean;
                if (var < 0.f) var = 0.f;
                float is = 1.0f / sqrtf(var + 1e-5f);
                float v = (orow[ir.off + c] - mean) * is;
                if (accumulate) drow[ir.off + c] += v; else drow[ir.off + c] = v;
            }
        } else {
            for (int c = 0; c < ir.m; ++c, ++ch) {
                float nn = stats[128 + ch] * invN;
                float is = 1.0f / sqrtf(nn + 1e-5f);
                int b3 = ir.off + c * 3;
                for (int k = 0; k < 3; ++k) {
                    float v = orow[b3 + k] * is;
                    if (accumulate) drow[b3 + k] += v; else drow[b3 + k] = v;
                }
            }
        }
    }
}

__global__ void pool_kernel(const float* __restrict__ pharm, const int* __restrict__ batch,
                            float* __restrict__ pooled, float* __restrict__ cnt)
{
    int t = blockIdx.x * blockDim.x + threadIdx.x;
    int n = t >> 5, c = t & 31;
    if (n >= NPHARM) return;
    int b = batch[n];
    atomicAdd(&pooled[b * NSC + c], pharm[(size_t)n * DMAX + c]);
    if (c == 0) atomicAdd(&cnt[b], 1.0f);
}

__global__ void final_kernel(const float* __restrict__ pooled, const float* __restrict__ cnt,
                             const float* __restrict__ w1, const float* __restrict__ w2,
                             float* __restrict__ out)
{
    int g = threadIdx.x;
    if (g >= NGRAPH) return;
    float inv = 1.0f / fmaxf(cnt[g], 1.0f);
    float p[NSC];
    for (int c = 0; c < NSC; ++c) p[c] = pooled[g * NSC + c] * inv;
    float acc = 0.f;
    for (int j = 0; j < NSC; ++j) {
        float h = 0.f;
        for (int c = 0; c < NSC; ++c) h += p[c] * w1[c * NSC + j];
        h = h > 0.f ? h : 0.f;
        acc += h * w2[j];
    }
    out[g] = acc;
}

// ---------------------------------------------------------------------------
// Host side
// ---------------------------------------------------------------------------
struct HIr { int m, l, p; };

static void build_layer(const HIr* in, int ni, const HIr* out, int no,
                        TPParams& tp, BNParams& bn, int& wn, int& dinv, int& doutv)
{
    HIr sh2[2] = {{1,0,1},{1,1,-1}};
    int ioff[5]; ioff[0] = 0;
    for (int i = 0; i < ni; ++i) ioff[i+1] = ioff[i] + in[i].m * (2*in[i].l + 1);
    int ooff[5]; ooff[0] = 0;
    for (int i = 0; i < no; ++i) ooff[i+1] = ooff[i] + out[i].m * (2*out[i].l + 1);
    dinv = ioff[ni]; doutv = ooff[no];
    int fan[4] = {0,0,0,0}; int pio[8];
    tp.np = 0; int woff = 0;
    for (int i1 = 0; i1 < ni; ++i1)
        for (int i2 = 0; i2 < 2; ++i2)
            for (int io = 0; io < no; ++io) {
                int l1 = in[i1].l, l2 = sh2[i2].l, lo = out[io].l;
                int dl = l1 - l2; if (dl < 0) dl = -dl;
                if (out[io].p == in[i1].p * sh2[i2].p && dl <= lo && lo <= l1 + l2) {
                    TPPath& P = tp.p[tp.np];
                    P.ioff = ioff[i1]; P.m1 = in[i1].m; P.l1 = l1; P.l2 = l2;
                    P.ooff = ooff[io]; P.mo = out[io].m; P.lo = lo;
                    P.woff = woff; P.norm = 0.f;
                    pio[tp.np] = io;
                    woff += in[i1].m * out[io].m;
                    fan[io] += in[i1].m;
                    tp.np++;
                }
            }
    wn = woff;
    for (int p = 0; p < tp.np; ++p) tp.p[p].norm = 1.0f / sqrtf((float)fan[pio[p]]);
    bn.nir = no;
    for (int i = 0; i < no; ++i) { bn.ir[i].off = ooff[i]; bn.ir[i].m = out[i].m; bn.ir[i].l = out[i].l; }
}

// canonical (insertion-order) leaf indexing
enum {
    L_PROT_TABLE = 0, L_PROT_W, L_PROT_B,
    L_PHARM_W1, L_PHARM_B1, L_PHARM_W2, L_PHARM_B2,
    L_PPM_W1, L_PPM_B1, L_PPM_W2, L_PPM_B2,
    L_HHM_W1, L_HHM_B1, L_HHM_W2, L_HHM_B2,
    L_LAYER0 = 15,                 // + l*12 + conv*4 + {0:w1,1:b1,2:w2,3:b2}; conv 0=pp,1=rev,2=hh
    L_FINAL_W1 = 63, L_FINAL_W2 = 64,
    L_COUNT = 65
};

static void sorted_map(int* mp)   // jax tree_flatten (sorted-key) position -> canonical index
{
    int p = 0;
    mp[p++] = L_FINAL_W1; mp[p++] = L_FINAL_W2;
    mp[p++] = L_HHM_B1; mp[p++] = L_HHM_B2; mp[p++] = L_HHM_W1; mp[p++] = L_HHM_W2;
    const int convs[3] = {2, 0, 1};   // sorted: hh, pp, rev
    for (int l = 0; l < 4; ++l)
        for (int ci = 0; ci < 3; ++ci) {
            int base = L_LAYER0 + l*12 + convs[ci]*4;
            mp[p++] = base + 1; mp[p++] = base + 3; mp[p++] = base + 0; mp[p++] = base + 2;
        }
    mp[p++] = L_PHARM_B1; mp[p++] = L_PHARM_B2; mp[p++] = L_PHARM_W1; mp[p++] = L_PHARM_W2;
    mp[p++] = L_PPM_B1; mp[p++] = L_PPM_B2; mp[p++] = L_PPM_W1; mp[p++] = L_PPM_W2;
    mp[p++] = L_PROT_B; mp[p++] = L_PROT_TABLE; mp[p++] = L_PROT_W;
}

static void canon_sizes(const int* wn, size_t* sz)
{
    sz[L_PROT_TABLE] = 14*32; sz[L_PROT_W] = 32*32; sz[L_PROT_B] = 32;
    sz[L_PHARM_W1] = 32*32; sz[L_PHARM_B1] = 32; sz[L_PHARM_W2] = 32*32; sz[L_PHARM_B2] = 32;
    sz[L_PPM_W1] = 50*32; sz[L_PPM_B1] = 32; sz[L_PPM_W2] = 32*32; sz[L_PPM_B2] = 32;
    sz[L_HHM_W1] = 50*32; sz[L_HHM_B1] = 32; sz[L_HHM_W2] = 32*32; sz[L_HHM_B2] = 32;
    for (int l = 0; l < 4; ++l)
        for (int c = 0; c < 3; ++c) {
            int base = L_LAYER0 + l*12 + c*4;
            sz[base+0] = 96*96; sz[base+1] = 96; sz[base+2] = (size_t)96*wn[l]; sz[base+3] = wn[l];
        }
    sz[L_FINAL_W1] = 32*32; sz[L_FINAL_W2] = 32;
}

extern "C" void kernel_launch(void* const* d_in, const int* in_sizes, int n_in,
                              void* d_out, int out_size, void* d_ws, size_t ws_size,
                              hipStream_t stream)
{
    (void)out_size;
    if (n_in < 8) return;

    // ---- irrep / tensor-product layer metadata ----
    static const HIr S0[1] = {{32,0,1}};
    static const HIr S1[2] = {{32,0,1},{8,1,-1}};
    static const HIr S2[3] = {{32,0,1},{8,1,-1},{8,1,1}};
    static const HIr S3[4] = {{32,0,1},{8,1,-1},{8,1,1},{32,0,-1}};
    static const HIr S4[1] = {{32,0,1}};
    const HIr* inir[4]  = {S0, S1, S2, S3}; int nin4[4]  = {1,2,3,4};
    const HIr* outir[4] = {S1, S2, S3, S4}; int nout4[4] = {2,3,4,1};
    TPParams tp[4]; BNParams bn[4]; int wn[4], din[4], dout[4];
    for (int l = 0; l < 4; ++l)
        build_layer(inir[l], nin4[l], outir[l], nout4[l], tp[l], bn[l], wn[l], din[l], dout[l]);
    // wn = {1280, 1664, 2048, 1280}

    // ---- map input leaves ----
    const float* leaf[L_COUNT];
    const float* x_pharm = (const float*)d_in[0];
    const float* pp_attr = (const float*)d_in[1];
    const float* hh_attr = (const float*)d_in[2];
    const int *x_prot, *pp_ei, *hh_ei, *batch;
    if (n_in >= 72) {
        if (in_sizes[3] == 14*32) {          // insertion-order leaves
            for (int k = 0; k < L_COUNT; ++k) leaf[k] = (const float*)d_in[3 + k];
        } else {                             // jax sorted-key order
            int mp[L_COUNT]; sorted_map(mp);
            for (int p = 0; p < L_COUNT; ++p) leaf[mp[p]] = (const float*)d_in[3 + p];
        }
        x_prot = (const int*)d_in[68]; pp_ei = (const int*)d_in[69];
        hh_ei  = (const int*)d_in[70]; batch = (const int*)d_in[71];
    } else {                                 // params concatenated into one buffer
        const float* base = (const float*)d_in[3];
        int mp[L_COUNT]; sorted_map(mp);
        size_t sz[L_COUNT]; canon_sizes(wn, sz);
        size_t off = 0;
        for (int p = 0; p < L_COUNT; ++p) { leaf[mp[p]] = base + off; off += sz[mp[p]]; }
        x_prot = (const int*)d_in[4]; pp_ei = (const int*)d_in[5];
        hh_ei  = (const int*)d_in[6]; batch = (const int*)d_in[7];
    }
    const int* pp_src = pp_ei;            const int* pp_dst = pp_ei + EPP;
    const int* hh_src = hh_ei;            const int* hh_dst = hh_ei + EHH;

    // ---- workspace carve (floats) ----
    float* W = (float*)d_ws;
    size_t off = 0;
    auto carve = [&](size_t n) { float* p = W + off; off += n; return p; };
    float* protA  = carve((size_t)NPROT * DMAX);
    float* protB  = carve((size_t)NPROT * DMAX);
    float* pharmA = carve((size_t)NPHARM * DMAX);
    float* pharmB = carve((size_t)NPHARM * DMAX);
    float* obuf   = carve((size_t)NPROT * DMAX);
    float* e_pp   = carve((size_t)EPP * NSC);
    float* e_hh   = carve((size_t)EHH * NSC);
    float* sh_pp  = carve((size_t)EPP * 4);
    float* sh_rv  = carve((size_t)EPP * 4);
    float* sh_hh  = carve((size_t)EHH * 4);
    float* sm_pp  = carve((size_t)EPP * RDIMC);
    float* sm_hh  = carve((size_t)EHH * RDIMC);
    float* tmp    = carve((size_t)EPP * NSC);
    float* stats  = carve(256);
    float* pooled = carve((size_t)NGRAPH * NSC);
    float* cnt    = carve(NGRAPH);
    size_t wsf = ws_size / sizeof(float);
    size_t remain = wsf > off ? wsf - off : 0;
    const size_t perEdge = 96 + 96 + 2048;
    long long cel = (long long)(remain / perEdge);
    int CE = cel > EPP ? EPP : (int)cel;
    if (CE < 1) CE = 1;
    float* pf = carve((size_t)CE * 96);
    float* hb = carve((size_t)CE * 96);
    float* wc = carve((size_t)CE * 2048);

    auto zero = [&](float* p, size_t n) {
        size_t blocks = (n + 255) / 256; if (blocks > 8192) blocks = 8192;
        fill_kernel<<<dim3((unsigned)blocks), dim3(256), 0, stream>>>(p, n);
    };
    auto gemm = [&](const float* A, int lda, const float* B, const float* bias,
                    float* C, int ldc, int M, int N, int K,
                    const int* gather, int preRelu, int reluOut) {
        int nsub = (N >= 128) ? 2 : 1;
        int colspan = 64 * nsub;
        dim3 g((M + 15) / 16, (N + colspan - 1) / colspan), b(128);
        wmma_gemm_kernel<<<g, b, 0, stream>>>(A, lda, B, bias, C, ldc, M, N, K,
                                              gather, preRelu, reluOut, nsub);
    };

    // ---- embeddings ----
    zero(protA, (size_t)NPROT * DMAX);
    zero(pharmA, (size_t)NPHARM * DMAX);
    gemm(leaf[L_PROT_TABLE], 32, leaf[L_PROT_W], leaf[L_PROT_B],
         protA, DMAX, NPROT, 32, 32, x_prot, 1, 0);
    gemm(x_pharm, 32, leaf[L_PHARM_W1], leaf[L_PHARM_B1], tmp, 32, NPHARM, 32, 32, nullptr, 0, 1);
    gemm(tmp, 32, leaf[L_PHARM_W2], leaf[L_PHARM_B2], pharmA, DMAX, NPHARM, 32, 32, nullptr, 0, 0);

    // ---- edge geometry + edge MLPs ----
    edge_geom_kernel<<<(EPP + 63) / 64, 64, 0, stream>>>(pp_attr, EPP, sh_pp, sh_rv, sm_pp);
    edge_geom_kernel<<<(EHH + 63) / 64, 64, 0, stream>>>(hh_attr, EHH, sh_hh, nullptr, sm_hh);
    gemm(sm_pp, RDIMC, leaf[L_PPM_W1], leaf[L_PPM_B1], tmp, 32, EPP, 32, RDIMC, nullptr, 0, 1);
    gemm(tmp, 32, leaf[L_PPM_W2], leaf[L_PPM_B2], e_pp, 32, EPP, 32, 32, nullptr, 0, 0);
    gemm(sm_hh, RDIMC, leaf[L_HHM_W1], leaf[L_HHM_B1], tmp, 32, EHH, 32, RDIMC, nullptr, 0, 1);
    gemm(tmp, 32, leaf[L_HHM_W2], leaf[L_HHM_B2], e_hh, 32, EHH, 32, 32, nullptr, 0, 0);

    float* curP = protA; float* newP = protB;
    float* curH = pharmA; float* newH = pharmB;

    auto run_conv = [&](const float* efeat, int E,
                        const float* fA, const int* iA, const float* fB, const int* iB,
                        const float* xsrc, const int* sidx, const int* didx,
                        const float* sh, int Ndst, const float* res,
                        int l, int conv, float* target, int accumulate) {
        int base = L_LAYER0 + l * 12 + conv * 4;
        const float* W1 = leaf[base + 0]; const float* b1 = leaf[base + 1];
        const float* W2 = leaf[base + 2]; const float* b2 = leaf[base + 3];
        zero(obuf, (size_t)Ndst * DMAX);
        for (int e0 = 0; e0 < E; e0 += CE) {
            int ce = (E - e0 < CE) ? (E - e0) : CE;
            build_feat_kernel<<<(ce * 32 + 255) / 256, 256, 0, stream>>>(
                efeat, fA, iA, fB, iB, pf, e0, ce);
            gemm(pf, 96, W1, b1, hb, 96, ce, 96, 96, nullptr, 0, 1);
            gemm(hb, 96, W2, b2, wc, wn[l], ce, wn[l], 96, nullptr, 0, 0);
            tp_scatter_kernel<<<(ce + 7) / 8, 256, 0, stream>>>(
                xsrc, sidx, didx, sh, wc, obuf, wn[l], e0, ce, tp[l]);
        }
        int cmin = din[l] < dout[l] ? din[l] : dout[l];
        add_res_kernel<<<((size_t)Ndst * cmin + 255) / 256, 256, 0, stream>>>(obuf, res, Ndst, cmin);
        int nch = 0;
        for (int i = 0; i < bn[l].nir; ++i) nch += bn[l].ir[i].m;
        bn_reduce_kernel<<<nch, 256, 0, stream>>>(obuf, Ndst, bn[l], stats);
        bn_apply_kernel<<<(Ndst + 255) / 256, 256, 0, stream>>>(
            obuf, Ndst, bn[l], stats, target, dout[l], accumulate);
    };

    // ---- 4 equivariant message-passing layers ----
    for (int l = 0; l < 4; ++l) {
        // pp: prot -> pharm (overwrite new pharm)
        run_conv(e_pp, EPP, curP, pp_src, curH, pp_dst,
                 curP, pp_src, pp_dst, sh_pp, NPHARM, curH, l, /*pp*/0, newH, 0);
        // hh: pharm -> pharm (accumulate into new pharm)
        run_conv(e_hh, EHH, curH, hh_src, curH, hh_dst,
                 curH, hh_src, hh_dst, sh_hh, NPHARM, curH, l, /*hh*/2, newH, 1);
        // rev: pharm -> prot (overwrite new prot)
        run_conv(e_pp, EPP, curH, pp_dst, curP, pp_src,
                 curH, pp_dst, pp_src, sh_rv, NPROT, curP, l, /*rev*/1, newP, 0);
        float* t0 = curP; curP = newP; newP = t0;
        float* t1 = curH; curH = newH; newH = t1;
    }

    // ---- mean pool over graphs + final MLP ----
    zero(pooled, (size_t)NGRAPH * NSC);
    zero(cnt, NGRAPH);
    pool_kernel<<<(NPHARM * 32 + 255) / 256, 256, 0, stream>>>(curH, batch, pooled, cnt);
    final_kernel<<<1, 64, 0, stream>>>(pooled, cnt, leaf[L_FINAL_W1], leaf[L_FINAL_W2],
                                       (float*)d_out);
}
